// STPAttention_36472862277838
// MI455X (gfx1250) — compile-verified, matrix operand order checked by reading, not run on hardware
//
#include <hip/hip_runtime.h>
#include <hip/hip_bf16.h>
#include <stdint.h>

// ---------------- problem constants (match reference) ----------------
#define BB 4
#define LL 2048
#define DD 1024
#define HH 16
#define HD 64   // head dim d

typedef __bf16 bf16_t;
typedef __attribute__((ext_vector_type(16))) __bf16 v16bf;
typedef __attribute__((ext_vector_type(8)))  float  v8f;

union FragBf { v16bf v; unsigned int u[8]; };
union F4x4   { float4 v[4]; float f[16]; };

#if defined(__has_builtin)
#if __has_builtin(__builtin_amdgcn_tensor_load_to_lds) && __has_builtin(__builtin_amdgcn_s_wait_tensorcnt)
#define USE_TDM 1
#endif
#endif
#ifndef USE_TDM
#define USE_TDM 0
#endif

#if USE_TDM
typedef __attribute__((ext_vector_type(4))) unsigned int tdm_v4u;
typedef __attribute__((ext_vector_type(8))) int          tdm_v8i;
typedef __attribute__((ext_vector_type(4))) int          tdm_v4i;

// Issue one 2D TDM tile load (bf16 elements) Global -> LDS.
// LDS padding: 1 DWORD inserted every 16 DWORDs => LDS row stride = 17 dwords
// for a 32-element (64B) bf16 tile row — matches load_frag_lds addressing.
// 6-arg builtin form: (g0 u32x4, g1 i32x8, g2 i32x4, g3 i32x4, i32x8, cpol);
// groups 2/3 and the trailing i32x8 are zero (2D tile, no iterate/gather).
__device__ inline void tdm_load_2d(unsigned lds_addr, const void* gaddr,
                                   unsigned tensor_d0, unsigned tensor_d1,
                                   unsigned stride0_elems,
                                   unsigned tile_d0, unsigned tile_d1)
{
  const unsigned long long ga = (unsigned long long)(uintptr_t)gaddr;
  tdm_v4u g0;
  g0.x = 0x1u;                                   // count=1, user descriptor
  g0.y = lds_addr;                               // LDS byte address
  g0.z = (unsigned)(ga & 0xFFFFFFFFu);           // global_addr[31:0]
  g0.w = (unsigned)((ga >> 32) & 0x01FFFFFFu)    // global_addr[56:32]
       | (2u << 30);                             // type = 2 (image)
  tdm_v8i g1;
  g1[0] = (int)((1u << 16)                       // data_size = 2 bytes
              | (1u << 20)                       // pad_enable
              | (3u << 22));                     // pad_interval: 16 DWORDs
                                                 // pad_amount[31:25]=0 -> 1 DWORD
  g1[1] = (int)((tensor_d0 & 0xFFFFu) << 16);    // tensor_dim0[15:0]
  g1[2] = (int)(((tensor_d0 >> 16) & 0xFFFFu)
              | ((tensor_d1 & 0xFFFFu) << 16));  // tensor_dim0[31:16] | tensor_dim1[15:0]
  g1[3] = (int)(((tensor_d1 >> 16) & 0xFFFFu)
              | ((tile_d0 & 0xFFFFu) << 16));    // tensor_dim1[31:16] | tile_dim0
  g1[4] = (int)(tile_d1 & 0xFFFFu);              // tile_dim1 (tile_dim2 = 0)
  g1[5] = (int)stride0_elems;                    // tensor_dim0_stride[31:0]
  g1[6] = 0;
  g1[7] = 0;
  tdm_v4i g2 = {0, 0, 0, 0};
  tdm_v4i g3 = {0, 0, 0, 0};
  tdm_v8i g4 = {0, 0, 0, 0, 0, 0, 0, 0};
  __builtin_amdgcn_tensor_load_to_lds(g0, g1, g2, g3, g4, 0);
}
#endif

// Load one 16x32 bf16 operand fragment from an LDS tile stored as
// row-major uints with row stride 17 uints (34 bf16, padded).
// Layout per CDNA5 ISA 7.12.2 (16-bit A matrix):
//   lane 0-15 : rows 0-15, K = {0..7, 16..23}
//   lane 16-31: rows 0-15, K = {8..15, 24..31}
__device__ inline v16bf load_frag_lds(const unsigned int* tile_u, int lane) {
  const int r  = lane & 15;
  const int kh = (lane >> 4) & 1;
  const unsigned int* p = tile_u + r * 17;
  FragBf f;
#pragma unroll
  for (int j = 0; j < 8; ++j) {
    const int ku = ((j >> 2) << 3) + kh * 4 + (j & 3);
    f.u[j] = p[ku];
  }
  return f.v;
}

// ---------------- f32 -> bf16 conversion ----------------
__global__ __launch_bounds__(256) void f32_to_bf16(const float* __restrict__ in,
                                                   bf16_t* __restrict__ out, int n) {
  int i = blockIdx.x * blockDim.x + threadIdx.x;
  if (i < n) out[i] = (bf16_t)in[i];
}

// ---------------- bf16 WMMA GEMM:  C[M,N] = A[M,K] * W[N,K]^T + bias[N] ----------------
// Block tile 128(M) x 64(N), K step 32; 8 waves x (32x32 per wave).
// TDM path: wave 0 DMAs tiles into double-buffered LDS; TENSORcnt + one
// workgroup barrier per k-step; TDM(k+1) overlaps WMMA(k).
__global__ __launch_bounds__(256) void gemm_bf16_wmma(
    const bf16_t* __restrict__ A, const bf16_t* __restrict__ W,
    const float* __restrict__ bias, float* __restrict__ C,
    int M, int N, int K)
{
#if USE_TDM
  __shared__ unsigned int As_u[2][128 * 17];
  __shared__ unsigned int Bs_u[2][64 * 17];
#else
  __shared__ unsigned int As_u[1][128 * 17];
  __shared__ unsigned int Bs_u[1][64 * 17];
#endif

  const int tid  = threadIdx.x;
  const int lane = tid & 31;
  const int wid  = tid >> 5;
  const int wm   = wid & 3;   // 4 waves along M (32 rows each)
  const int wn   = wid >> 2;  // 2 waves along N (32 cols each)
  const int m0   = blockIdx.y * 128;
  const int n0   = blockIdx.x * 64;

  v8f acc[2][2];
#pragma unroll
  for (int i = 0; i < 2; ++i)
#pragma unroll
    for (int j = 0; j < 2; ++j) acc[i][j] = v8f{};

#if USE_TDM
  const unsigned ldsA[2] = { (unsigned)(uintptr_t)&As_u[0][0],
                             (unsigned)(uintptr_t)&As_u[1][0] };
  const unsigned ldsB[2] = { (unsigned)(uintptr_t)&Bs_u[0][0],
                             (unsigned)(uintptr_t)&Bs_u[1][0] };
  const int nK = K >> 5;

  if (wid == 0) { // preload k-tile 0 into buffer 0
    tdm_load_2d(ldsA[0], (const char*)A + 2 * ((size_t)m0 * K), K, 128, K, 32, 128);
    tdm_load_2d(ldsB[0], (const char*)W + 2 * ((size_t)n0 * K), K, 64,  K, 32, 64);
  }

  for (int kt = 0; kt < nK; ++kt) {
    const int cur = kt & 1;
    if (wid == 0)
      __builtin_amdgcn_s_wait_tensorcnt(0);   // buffer 'cur' DMA complete
    __syncthreads();                          // visibility + prev readers done
    if (wid == 0 && kt + 1 < nK) {            // overlap next DMA with WMMA
      const size_t k1 = (size_t)(kt + 1) << 5;
      tdm_load_2d(ldsA[cur ^ 1], (const char*)A + 2 * ((size_t)m0 * K + k1), K, 128, K, 32, 128);
      tdm_load_2d(ldsB[cur ^ 1], (const char*)W + 2 * ((size_t)n0 * K + k1), K, 64,  K, 32, 64);
    }

    v16bf a0 = load_frag_lds(&As_u[cur][(wm * 32 +  0) * 17], lane);
    v16bf a1 = load_frag_lds(&As_u[cur][(wm * 32 + 16) * 17], lane);
    v16bf b0 = load_frag_lds(&Bs_u[cur][(wn * 32 +  0) * 17], lane);
    v16bf b1 = load_frag_lds(&Bs_u[cur][(wn * 32 + 16) * 17], lane);

    acc[0][0] = __builtin_amdgcn_wmma_f32_16x16x32_bf16(false, a0, false, b0, (short)0, acc[0][0], false, false);
    acc[0][1] = __builtin_amdgcn_wmma_f32_16x16x32_bf16(false, a0, false, b1, (short)0, acc[0][1], false, false);
    acc[1][0] = __builtin_amdgcn_wmma_f32_16x16x32_bf16(false, a1, false, b0, (short)0, acc[1][0], false, false);
    acc[1][1] = __builtin_amdgcn_wmma_f32_16x16x32_bf16(false, a1, false, b1, (short)0, acc[1][1], false, false);
  }
#else
  const unsigned int* Au = (const unsigned int*)A;
  const unsigned int* Wu = (const unsigned int*)W;
  const int Ku = K >> 1;

  const int arow  = tid >> 1;
  const int ahalf = tid & 1;
  const int brow  = tid >> 2;
  const int bq    = tid & 3;

  for (int k0 = 0; k0 < K; k0 += 32) {
    if (k0 + 32 < K)
      __builtin_prefetch(&A[(size_t)(m0 + arow) * K + k0 + 32], 0, 0);

    const int ku0 = k0 >> 1;
    {
      const size_t src = (size_t)(m0 + arow) * Ku + ku0 + ahalf * 8;
      unsigned int* dst = &As_u[0][arow * 17 + ahalf * 8];
#pragma unroll
      for (int j = 0; j < 8; ++j) dst[j] = Au[src + j];
    }
    {
      const size_t src = (size_t)(n0 + brow) * Ku + ku0 + bq * 4;
      unsigned int* dst = &Bs_u[0][brow * 17 + bq * 4];
#pragma unroll
      for (int j = 0; j < 4; ++j) dst[j] = Wu[src + j];
    }
    __syncthreads();

    v16bf a0 = load_frag_lds(&As_u[0][(wm * 32 +  0) * 17], lane);
    v16bf a1 = load_frag_lds(&As_u[0][(wm * 32 + 16) * 17], lane);
    v16bf b0 = load_frag_lds(&Bs_u[0][(wn * 32 +  0) * 17], lane);
    v16bf b1 = load_frag_lds(&Bs_u[0][(wn * 32 + 16) * 17], lane);

    acc[0][0] = __builtin_amdgcn_wmma_f32_16x16x32_bf16(false, a0, false, b0, (short)0, acc[0][0], false, false);
    acc[0][1] = __builtin_amdgcn_wmma_f32_16x16x32_bf16(false, a0, false, b1, (short)0, acc[0][1], false, false);
    acc[1][0] = __builtin_amdgcn_wmma_f32_16x16x32_bf16(false, a1, false, b0, (short)0, acc[1][0], false, false);
    acc[1][1] = __builtin_amdgcn_wmma_f32_16x16x32_bf16(false, a1, false, b1, (short)0, acc[1][1], false, false);

    __syncthreads();
  }
#endif

  // C/D layout: VGPR r -> M = r + (lane/16)*8 ; N = lane%16
  const int col = lane & 15;
  const int rb  = (lane >> 4) << 3;
#pragma unroll
  for (int i = 0; i < 2; ++i)
#pragma unroll
    for (int j = 0; j < 2; ++j) {
      const int m = m0 + wm * 32 + i * 16 + rb;
      const int n = n0 + wn * 32 + j * 16 + col;
      const float bv = bias ? bias[n] : 0.0f;
#pragma unroll
      for (int r = 0; r < 8; ++r)
        C[(size_t)(m + r) * N + n] = acc[i][j][r] + bv;
    }
}

// ---------------- sequential retention scan ----------------
// grid = B*H blocks, 256 threads (8 waves). Each wave owns 8 state rows;
// each lane owns 16 state elements. Barrier-free: 512 independent chains.
__global__ __launch_bounds__(256) void scan_kernel(
    const float* __restrict__ qkv,        // (B,L,3,H,d): [(b*L+t)*3D + s*D + h*d + i]
    const float* __restrict__ Lambda_raw, // (H,d,d)
    const float* __restrict__ Gamma,      // (H,d,d)
    const float* __restrict__ W_static,   // (H,d,d)
    float* __restrict__ y)                // (B,L,D)
{
  const int bh   = blockIdx.x;
  const int b    = bh / HH;
  const int h    = bh % HH;
  const int tid  = threadIdx.x;
  const int lane = tid & 31;
  const int wave = tid >> 5;
  const int row  = wave * 8 + (lane >> 2); // 0..63
  const int c0   = (lane & 3) << 4;        // 0,16,32,48
  const float scale = 0.125f;              // 1/sqrt(64)

  float ret[16], gam[16], wst[16], s[16];
  const size_t hoff = (size_t)h * HD * HD + (size_t)row * HD + c0;
#pragma unroll
  for (int j = 0; j < 16; ++j) {
    const float lam = Lambda_raw[hoff + j];
    const float sig = 1.0f / (1.0f + __expf(-lam));
    ret[j] = 1.0f - sig;
    gam[j] = Gamma[hoff + j];
    wst[j] = W_static[hoff + j];
    s[j]   = 0.0f;
  }

  const size_t strideT = 3 * DD;
  const float* base = qkv + (size_t)b * LL * strideT + (size_t)h * HD;
  float* yb = y + (size_t)b * LL * DD + (size_t)h * HD + row;

  for (int t = 0; t < LL; ++t) {
    const float* p = base + (size_t)t * strideT;
    const float4* qv = (const float4*)(p + c0);
    const float4* kv = (const float4*)(p + DD + c0);
    F4x4 qf, kf;
#pragma unroll
    for (int j = 0; j < 4; ++j) { qf.v[j] = qv[j]; kf.v[j] = kv[j]; }
    const float vr = p[2 * DD + row];

    float pr = 0.0f;
#pragma unroll
    for (int j = 0; j < 16; ++j) {
      const float kk = kf.f[j] * scale;
      s[j] = ret[j] * s[j] + gam[j] * (vr * kk);
      pr = __builtin_fmaf(wst[j] + s[j], qf.f[j], pr);
    }
    pr += __shfl_xor(pr, 1, 32);
    pr += __shfl_xor(pr, 2, 32);
    if ((lane & 3) == 0) yb[(size_t)t * DD] = pr;
  }
}

// ---------------- launcher ----------------
extern "C" void kernel_launch(void* const* d_in, const int* in_sizes, int n_in,
                              void* d_out, int out_size, void* d_ws, size_t ws_size,
                              hipStream_t stream) {
  const float* x          = (const float*)d_in[0];
  const float* Wqkv_w     = (const float*)d_in[1];
  const float* Wqkv_b     = (const float*)d_in[2];
  const float* out_w      = (const float*)d_in[3];
  const float* out_b      = (const float*)d_in[4];
  const float* W_static   = (const float*)d_in[5];
  const float* Lambda_raw = (const float*)d_in[6];
  const float* Gamma      = (const float*)d_in[7];

  const size_t nX    = (size_t)BB * LL * DD;     // 8,388,608
  const size_t nWqkv = (size_t)3 * DD * DD;      // 3,145,728
  const size_t nOutW = (size_t)DD * DD;          // 1,048,576
  const size_t nQKV  = (size_t)BB * LL * 3 * DD; // 25,165,824

  char* w = (char*)d_ws;
  bf16_t* xb    = (bf16_t*)w; w += nX    * 2;
  bf16_t* wqkvb = (bf16_t*)w; w += nWqkv * 2;
  bf16_t* outwb = (bf16_t*)w; w += nOutW * 2;
  bf16_t* ybf   = (bf16_t*)w; w += nX    * 2;
  float*  qkvf  = (float*)w;  w += nQKV  * 4;
  float*  yf    = (float*)w;  w += nX    * 4;

  float* out = (float*)d_out;
  const int tpb = 256;

  f32_to_bf16<<<(unsigned)((nX    + tpb - 1) / tpb), tpb, 0, stream>>>(x,      xb,    (int)nX);
  f32_to_bf16<<<(unsigned)((nWqkv + tpb - 1) / tpb), tpb, 0, stream>>>(Wqkv_w, wqkvb, (int)nWqkv);
  f32_to_bf16<<<(unsigned)((nOutW + tpb - 1) / tpb), tpb, 0, stream>>>(out_w,  outwb, (int)nOutW);

  // QKV projection: M=8192, N=3072, K=1024
  dim3 g1(3 * DD / 64, BB * LL / 128);
  gemm_bf16_wmma<<<g1, 256, 0, stream>>>(xb, wqkvb, Wqkv_b, qkvf, BB * LL, 3 * DD, DD);

  // sequential retention scan
  scan_kernel<<<BB * HH, 256, 0, stream>>>(qkvf, Lambda_raw, Gamma, W_static, yf);

  // output projection: M=8192, N=1024, K=1024
  f32_to_bf16<<<(unsigned)((nX + tpb - 1) / tpb), tpb, 0, stream>>>(yf, ybf, (int)nX);
  dim3 g2(DD / 64, BB * LL / 128);
  gemm_bf16_wmma<<<g2, 256, 0, stream>>>(ybf, outwb, out_b, out, BB * LL, DD, DD);
}
